// MultiHeadAttention_2293512536942
// MI455X (gfx1250) — matrix operand order repaired and checked by reading, not compiled, not run on hardware
//
#include <hip/hip_runtime.h>
#include <hip/hip_bf16.h>

// ---------------------------------------------------------------------------
// MultiHeadAttention for MI455X (gfx1250): bf16 WMMA everywhere.
// B=4, N=2048, D=1024, H=16, hd=64.  Compute-bound (~137 GFLOP vs ~7us HBM),
// so all three GEMM stages use v_wmma_f32_16x16x32_bf16 (f32 accumulate).
// GEMM waves use 32x64 tiles: 8 WMMAs per 12x16B loads (~21 FLOP/B from L2).
// ---------------------------------------------------------------------------

#define Bb    4
#define NSEQ  2048
#define DIM   1024
#define NH    16
#define HD    64
#define D3    (3 * DIM)
#define MROWS (Bb * NSEQ)          // 8192
#define SCALE 0.125f               // 1/sqrt(64)

typedef __attribute__((ext_vector_type(16))) __bf16          v16bf;
typedef __attribute__((ext_vector_type(8)))  float           v8f;
typedef __attribute__((ext_vector_type(8)))  unsigned short  v8u16;

union Frag {                       // 16x32 bf16 A-frag or 32x16 bf16 B-frag
    struct { v8u16 lo, hi; } h;
    v16bf v;
};

__device__ __forceinline__ unsigned short f2bf(float f) {
    unsigned u = __builtin_bit_cast(unsigned, f);
    u += 0x7FFFu + ((u >> 16) & 1u);           // round-to-nearest-even
    return (unsigned short)(u >> 16);
}

__device__ __forceinline__ v8u16 ld8(const unsigned short* p) {
    return *(const v8u16*)p;                   // 16-byte aligned by layout
}

__device__ __forceinline__ float red_max16(float v) {
#pragma unroll
    for (int m = 1; m < 16; m <<= 1) v = fmaxf(v, __shfl_xor(v, m, 32));
    return v;                                  // reduces within 16-lane half
}
__device__ __forceinline__ float red_sum16(float v) {
#pragma unroll
    for (int m = 1; m < 16; m <<= 1) v += __shfl_xor(v, m, 32);
    return v;
}

// ---------------------------------------------------------------------------
// f32 -> bf16 convert
// ---------------------------------------------------------------------------
__global__ void cvt_f32_bf16(const float* __restrict__ in,
                             unsigned short* __restrict__ out, int n) {
    int i = blockIdx.x * blockDim.x + threadIdx.x;
    if (i < n) out[i] = f2bf(in[i]);
}

// ---------------------------------------------------------------------------
// Wave-level GEMM core: one wave computes a 32(M) x 64(N) tile.
//   C[m,n] = sum_k A[m,k] * W[n,k]      (i.e. A @ W^T)
// A: bf16 [.., K] row-major starting at row m0 ; W: bf16 [.., K] starting row n0
// Each B fragment feeds two WMMAs (two M sub-tiles) -> half the B traffic.
// ---------------------------------------------------------------------------
__device__ __forceinline__ void gemm32x64(const unsigned short* __restrict__ A,
                                          const unsigned short* __restrict__ W,
                                          int K, int lane, v8f acc[2][4]) {
    const int ml = lane & 15, g = lane >> 4;
    const unsigned short* arow0 = A + ml * K;          // M sub-tile 0 row
    const unsigned short* arow1 = A + (16 + ml) * K;   // M sub-tile 1 row
    const unsigned short* brow  = W + ml * K;          // W row (=B column)
    for (int k0 = 0; k0 < K; k0 += 32) {
        __builtin_prefetch(arow0 + k0 + 512, 0, 1);    // global_prefetch_b8
        __builtin_prefetch(arow1 + k0 + 512, 0, 1);
        Frag a0, a1;
        a0.h.lo = ld8(arow0 + k0 + g * 8);             // K = kb+0..7
        a0.h.hi = ld8(arow0 + k0 + g * 8 + 16);        // K = kb+16..23
        a1.h.lo = ld8(arow1 + k0 + g * 8);
        a1.h.hi = ld8(arow1 + k0 + g * 8 + 16);
#pragma unroll
        for (int nt = 0; nt < 4; ++nt) {
            const unsigned short* bp = brow + nt * 16 * K + k0 + g * 8;
            Frag b;
            b.h.lo = ld8(bp);                          // K = kb2+0..7
            b.h.hi = ld8(bp + 16);                     // K = kb2+16..23
            acc[0][nt] = __builtin_amdgcn_wmma_f32_16x16x32_bf16(
                false, a0.v, false, b.v, (short)0, acc[0][nt], false, false);
            acc[1][nt] = __builtin_amdgcn_wmma_f32_16x16x32_bf16(
                false, a1.v, false, b.v, (short)0, acc[1][nt], false, false);
        }
    }
}

// ---------------------------------------------------------------------------
// Stage 1: qkv = x @ qkv_w^T + qkv_b, scattered to Q,K [B,H,N,hd] and V^T [B,H,hd,N]
// grid = (MROWS/256, D3/64), block = 256 (8 waves, each a 32x64 tile)
// ---------------------------------------------------------------------------
__global__ void qkv_gemm(const unsigned short* __restrict__ xb,
                         const unsigned short* __restrict__ wq,
                         const float* __restrict__ bias,
                         unsigned short* __restrict__ qO,
                         unsigned short* __restrict__ kO,
                         unsigned short* __restrict__ vT) {
    const int lane = threadIdx.x & 31, wave = threadIdx.x >> 5;
    const int m0 = (blockIdx.x * 8 + wave) * 32;
    const int n0 = blockIdx.y * 64;
    v8f zero = {};
    v8f acc[2][4] = {{zero, zero, zero, zero}, {zero, zero, zero, zero}};
    gemm32x64(xb + m0 * DIM, wq + n0 * DIM, DIM, lane, acc);

    const int g = lane >> 4, ml = lane & 15;
#pragma unroll
    for (int mt = 0; mt < 2; ++mt) {
#pragma unroll
        for (int nt = 0; nt < 4; ++nt) {
            const int ncol  = n0 + nt * 16 + ml;
            const int which = ncol >> 10;              // 0=q 1=k 2=v
            const int rem   = ncol & 1023;
            const int h = rem >> 6, hd = rem & 63;
            const float bv = bias[ncol];
#pragma unroll
            for (int r = 0; r < 8; ++r) {
                const int m   = m0 + mt * 16 + r + 8 * g;   // global token row
                const int b   = m >> 11, seq = m & 2047;
                const unsigned short o = f2bf(acc[mt][nt][r] + bv);
                const int bh = b * NH + h;
                if (which == 0)      qO[(bh * NSEQ + seq) * HD + hd] = o;
                else if (which == 1) kO[(bh * NSEQ + seq) * HD + hd] = o;
                else                 vT[(bh * HD + hd) * NSEQ + seq] = o;
            }
        }
    }
}

// ---------------------------------------------------------------------------
// Stage 2: flash attention. grid = (NSEQ/128, B*H), block = 256.
// Each wave: 16 query rows, streams keys in chunks of 32, online softmax.
// (16 q-rows/wave keeps VGPRs ~<128: aq 16 + O 32 + softmax state 16 + frags.)
// ---------------------------------------------------------------------------
__global__ void flash_attn(const unsigned short* __restrict__ qI,
                           const unsigned short* __restrict__ kI,
                           const unsigned short* __restrict__ vT,
                           unsigned short* __restrict__ ao) {
    __shared__ __align__(64) unsigned short pst[8][16 * 32];  // P restage / wave

    const int lane = threadIdx.x & 31, wave = threadIdx.x >> 5;
    const int bh = blockIdx.y;                     // 0..63
    const int b = bh >> 4, h = bh & 15;
    const int q0 = (blockIdx.x * 8 + wave) * 16;
    const int g = lane >> 4, ml = lane & 15;

    const unsigned short* Q = qI + bh * NSEQ * HD;
    const unsigned short* K = kI + bh * NSEQ * HD;
    const unsigned short* V = vT + bh * HD * NSEQ;

    // Q A-fragments for d = 0..31 and 32..63 (resident for whole pass)
    Frag aq[2];
#pragma unroll
    for (int dd = 0; dd < 2; ++dd) {
        const unsigned short* qp = Q + (q0 + ml) * HD + dd * 32 + g * 8;
        aq[dd].h.lo = ld8(qp);
        aq[dd].h.hi = ld8(qp + 16);
    }

    v8f zero = {};
    v8f oacc[4] = {zero, zero, zero, zero};        // O: 16 x 64 (4 C-tiles)
    float mrow[8], lrow[8];
#pragma unroll
    for (int r = 0; r < 8; ++r) { mrow[r] = -1e30f; lrow[r] = 0.f; }

    unsigned short* ps = pst[wave];

    for (int kk = 0; kk < NSEQ; kk += 32) {
        if (kk + 32 < NSEQ)
            __builtin_prefetch(K + (kk + 32 + ml) * HD, 0, 1);

        // ---- S = scale * Q K^T for two 16-key sub-tiles ----
        v8f s[2];
#pragma unroll
        for (int kt = 0; kt < 2; ++kt) {
            v8f c = {};
#pragma unroll
            for (int dd = 0; dd < 2; ++dd) {
                const unsigned short* kp =
                    K + (kk + kt * 16 + ml) * HD + dd * 32 + g * 8;
                Frag bk;
                bk.h.lo = ld8(kp);
                bk.h.hi = ld8(kp + 16);
                c = __builtin_amdgcn_wmma_f32_16x16x32_bf16(
                    false, aq[dd].v, false, bk.v, (short)0, c, false, false);
            }
            s[kt] = c * SCALE;
        }

        // ---- online softmax (rows live in C-layout: row r+8*g, col ml) ----
        float p0[8], p1[8];
#pragma unroll
        for (int r = 0; r < 8; ++r) {
            float lm = red_max16(fmaxf(s[0][r], s[1][r]));
            float mn = fmaxf(mrow[r], lm);
            float al = __expf(mrow[r] - mn);
            p0[r] = __expf(s[0][r] - mn);
            p1[r] = __expf(s[1][r] - mn);
            float rs = red_sum16(p0[r] + p1[r]);
            lrow[r] = lrow[r] * al + rs;
            mrow[r] = mn;
            oacc[0][r] *= al; oacc[1][r] *= al;
            oacc[2][r] *= al; oacc[3][r] *= al;
        }

        // ---- restage P (C-layout) -> A-fragment layout via LDS ----
#pragma unroll
        for (int r = 0; r < 8; ++r) {
            const int mr = r + 8 * g;
            ps[mr * 32 + ml]      = f2bf(p0[r]);
            ps[mr * 32 + 16 + ml] = f2bf(p1[r]);
        }
        __syncthreads();
        Frag pa;
        pa.h.lo = *(const v8u16*)&ps[ml * 32 + g * 8];
        pa.h.hi = *(const v8u16*)&ps[ml * 32 + g * 8 + 16];

        // ---- O += P V  (V^T rows are contiguous in k) ----
#pragma unroll
        for (int dt = 0; dt < 4; ++dt) {
            const unsigned short* vp = V + (dt * 16 + ml) * NSEQ + kk + g * 8;
            Frag bv;
            bv.h.lo = ld8(vp);
            bv.h.hi = ld8(vp + 16);
            oacc[dt] = __builtin_amdgcn_wmma_f32_16x16x32_bf16(
                false, pa.v, false, bv.v, (short)0, oacc[dt], false, false);
        }
        __syncthreads();
    }

    // ---- normalize and store to [B*N, D] bf16 for the proj GEMM ----
#pragma unroll
    for (int r = 0; r < 8; ++r) {
        const float inv = 1.0f / lrow[r];
        const int m = q0 + r + 8 * g;
        const int row = (b * NSEQ + m) * DIM + h * HD;
#pragma unroll
        for (int dt = 0; dt < 4; ++dt)
            ao[row + dt * 16 + ml] = f2bf(oacc[dt][r] * inv);
    }
}

// ---------------------------------------------------------------------------
// Stage 3: out = attn @ proj_w^T + proj_b   (f32 output)
// grid = (MROWS/256, DIM/64), block = 256
// ---------------------------------------------------------------------------
__global__ void proj_gemm(const unsigned short* __restrict__ ao,
                          const unsigned short* __restrict__ wp,
                          const float* __restrict__ bias,
                          float* __restrict__ out) {
    const int lane = threadIdx.x & 31, wave = threadIdx.x >> 5;
    const int m0 = (blockIdx.x * 8 + wave) * 32;
    const int n0 = blockIdx.y * 64;
    v8f zero = {};
    v8f acc[2][4] = {{zero, zero, zero, zero}, {zero, zero, zero, zero}};
    gemm32x64(ao + m0 * DIM, wp + n0 * DIM, DIM, lane, acc);

    const int g = lane >> 4, ml = lane & 15;
#pragma unroll
    for (int mt = 0; mt < 2; ++mt) {
#pragma unroll
        for (int nt = 0; nt < 4; ++nt) {
            const int ncol = n0 + nt * 16 + ml;
            const float bv = bias[ncol];
#pragma unroll
            for (int r = 0; r < 8; ++r) {
                const int m = m0 + mt * 16 + r + 8 * g;
                out[m * DIM + ncol] = acc[mt][nt][r] + bv;
            }
        }
    }
}

// ---------------------------------------------------------------------------
// Host-side launch
// ---------------------------------------------------------------------------
extern "C" void kernel_launch(void* const* d_in, const int* in_sizes, int n_in,
                              void* d_out, int out_size, void* d_ws, size_t ws_size,
                              hipStream_t stream) {
    const float* x      = (const float*)d_in[0];
    const float* qkv_w  = (const float*)d_in[1];
    const float* qkv_b  = (const float*)d_in[2];
    const float* proj_w = (const float*)d_in[3];
    const float* proj_b = (const float*)d_in[4];
    float* out = (float*)d_out;

    char* ws = (char*)d_ws;
    size_t off = 0;
    unsigned short* xb  = (unsigned short*)(ws + off); off += (size_t)MROWS * DIM * 2;   // 16 MB
    unsigned short* wqb = (unsigned short*)(ws + off); off += (size_t)D3 * DIM * 2;      //  6 MB
    unsigned short* wpb = (unsigned short*)(ws + off); off += (size_t)DIM * DIM * 2;     //  2 MB
    unsigned short* qB  = (unsigned short*)(ws + off); off += (size_t)MROWS * DIM * 2;   // 16 MB
    unsigned short* kB  = (unsigned short*)(ws + off); off += (size_t)MROWS * DIM * 2;   // 16 MB
    unsigned short* vT  = (unsigned short*)(ws + off); off += (size_t)MROWS * DIM * 2;   // 16 MB
    unsigned short* aoB = (unsigned short*)(ws + off); off += (size_t)MROWS * DIM * 2;   // 16 MB

    const int nx = MROWS * DIM, nw = D3 * DIM, np = DIM * DIM;
    cvt_f32_bf16<<<(nx + 255) / 256, 256, 0, stream>>>(x, xb, nx);
    cvt_f32_bf16<<<(nw + 255) / 256, 256, 0, stream>>>(qkv_w, wqb, nw);
    cvt_f32_bf16<<<(np + 255) / 256, 256, 0, stream>>>(proj_w, wpb, np);

    qkv_gemm<<<dim3(MROWS / 256, D3 / 64), 256, 0, stream>>>(xb, wqb, qkv_b, qB, kB, vT);
    flash_attn<<<dim3(NSEQ / 128, Bb * NH), 256, 0, stream>>>(qB, kB, vT, aoB);
    proj_gemm<<<dim3(MROWS / 256, DIM / 64), 256, 0, stream>>>(aoB, wpb, proj_b, out);
}
